// PaiNN_80839874445344
// MI455X (gfx1250) — compile-verified, matrix operand order checked by reading, not run on hardware
//
#include <hip/hip_runtime.h>
#include <hip/hip_bf16.h>
#include <math.h>

// ---------------- problem constants ----------------
#define NN 8192
#define EE 131072
#define HH 256
#define LL 3
#define RR 128
#define NG 64

#define INV_SQRT_2f 0.70710678118654752f
#define INV_SQRT_3f 0.57735026918962576f
#define INV_SQRT_Hf 0.0625f   // 1/sqrt(256)

// ---------------- WMMA types ----------------
typedef __attribute__((ext_vector_type(16))) __bf16 v16bf;
typedef __attribute__((ext_vector_type(8)))  float  v8f;
typedef __attribute__((ext_vector_type(4)))  float  f32x4;

union Frag { v16bf v; f32x4 q[2]; };

__device__ __forceinline__ float silu(float x) { return x / (1.0f + __expf(-x)); }

__device__ __forceinline__ void load_frag(Frag& f, const __bf16* row, int off) {
  f.q[0] = *(const f32x4*)(row + off);
  f.q[1] = *(const f32x4*)(row + off + 16);
}

// =====================================================================
// Generic bf16 WMMA GEMM:  C[M,Nc] = act(A[M,K] * Bt[Nc,K]^T + bias)
// A: bf16 row-major [M,K];  Bt: bf16 [Nc,K] (weights pre-transposed)
// Block tile: 32 rows x 256 cols; 8 waves = 2 (m) x 4 (n); each wave
// computes a 16x64 strip (4 acc tiles) so one A fragment feeds 4 WMMAs.
// grid = (M/32, ceil(Nc/256)), block = 256.
// =====================================================================
__global__ __launch_bounds__(256) void gemm_wmma_bf16(
    const __bf16* __restrict__ A, const __bf16* __restrict__ Bt,
    const float* __restrict__ bias, void* __restrict__ C,
    int M, int Ncols, int K, int act, int out_bf16)
{
  int wave = threadIdx.x >> 5;
  int lane = threadIdx.x & 31;
  int mtile = blockIdx.x * 32 + (wave >> 2) * 16;
  int nbase = blockIdx.y * 256 + (wave & 3) * 64;
  if (nbase >= Ncols) return;

  int r16   = lane & 15;
  int khalf = (lane >> 4) * 8;            // 0 or 8 per ISA A/B layout

  const __bf16* arow = A  + (size_t)(mtile + r16) * K;
  const __bf16* brow = Bt + (size_t)(nbase + r16) * K;

  v8f acc[4] = {};
  for (int k0 = 0; k0 < K; k0 += 32) {
    Frag a;
    load_frag(a, arow, k0 + khalf);
    Frag b[4];
#pragma unroll
    for (int t = 0; t < 4; ++t)
      load_frag(b[t], brow + (size_t)t * 16 * K, k0 + khalf);
#pragma unroll
    for (int t = 0; t < 4; ++t)
      acc[t] = __builtin_amdgcn_wmma_f32_16x16x32_bf16(false, a.v, false, b[t].v,
                                                       (short)0, acc[t], false, false);
  }

  int rb = mtile + ((lane >> 4) * 8);     // C layout: lanes>=16 hold rows +8
#pragma unroll
  for (int t = 0; t < 4; ++t) {
    int col  = nbase + t * 16 + r16;
    float bs = bias ? bias[col] : 0.0f;
#pragma unroll
    for (int v = 0; v < 8; ++v) {
      float r = acc[t][v] + bs;
      if (act) r = silu(r);
      size_t off = (size_t)(rb + v) * Ncols + col;
      if (out_bf16) ((__bf16*)C)[off] = (__bf16)r;
      else          ((float*)C)[off]  = r;
    }
  }
}

// =====================================================================
// Fused edge kernel: per 16-edge tile, WMMA-compute rbfh = rbf @ rbf_w + b
// into LDS (never hits HBM), then message math + atomic scatter-add.
// A fragments (16x128 rbf tile, K=128 -> 4 frags) hoisted and reused
// across all 6 n-tiles per wave (24 WMMAs per A load).
// grid = E/16, block = 256 (8 waves x 6 n-tiles = 48 tiles of 16 cols)
// =====================================================================
__global__ __launch_bounds__(256) void edge_message(
    const __bf16* __restrict__ rbf,     // [E,128] bf16
    const __bf16* __restrict__ rbfwT,   // [768,128] bf16 (layer slice)
    const float*  __restrict__ rbf_b,   // [768]
    const float*  __restrict__ xh,      // [N,768]
    const float*  __restrict__ vec,     // [N,3,256]
    const float*  __restrict__ evec,    // [E,3]
    const int*    __restrict__ src, const int* __restrict__ dst,
    float* __restrict__ dx, float* __restrict__ dvec)
{
  __shared__ float rt[16][768 + 8];

  int wave = threadIdx.x >> 5;
  int lane = threadIdx.x & 31;
  int e0   = blockIdx.x * 16;
  int r16  = lane & 15;
  int khalf = (lane >> 4) * 8;

  const __bf16* arow = rbf + (size_t)(e0 + r16) * RR;
  Frag a[4];
#pragma unroll
  for (int k = 0; k < 4; ++k) load_frag(a[k], arow, k * 32 + khalf);

  int rb = (lane >> 4) * 8;
  for (int t = 0; t < 6; ++t) {
    int nb = (wave * 6 + t) * 16;
    const __bf16* brow = rbfwT + (size_t)(nb + r16) * RR;
    v8f acc = {};
#pragma unroll
    for (int k = 0; k < 4; ++k) {
      Frag b;
      load_frag(b, brow, k * 32 + khalf);
      acc = __builtin_amdgcn_wmma_f32_16x16x32_bf16(false, a[k].v, false, b.v,
                                                    (short)0, acc, false, false);
    }
    int col = nb + r16;
    float bs = rbf_b[col];
#pragma unroll
    for (int v = 0; v < 8; ++v) rt[rb + v][col] = acc[v] + bs;
  }
  __syncthreads();

  // phase 2: message + scatter (16 edges * 256 h-values)
  for (int i = threadIdx.x; i < 16 * HH; i += 256) {
    int el = i >> 8, h = i & 255;
    int e = e0 + el;
    int s = src[e], d = dst[e];
    float r1 = rt[el][h], r2 = rt[el][HH + h], r3 = rt[el][2 * HH + h];
    const float* xs = xh + (size_t)s * (3 * HH);
    float mx = xs[h] * r1;
    float m2 = xs[HH + h] * r2 * INV_SQRT_3f;
    float m3 = xs[2 * HH + h] * r3;
    atomicAdd(dx + (size_t)d * HH + h, mx);
#pragma unroll
    for (int dd = 0; dd < 3; ++dd) {
      float mv = (vec[((size_t)s * 3 + dd) * HH + h] * m2 +
                  m3 * evec[(size_t)e * 3 + dd]) * INV_SQRT_Hf;
      atomicAdd(dvec + ((size_t)d * 3 + dd) * HH + h, mv);
    }
  }
}

// ---------------- small helper kernels ----------------
__global__ void edge_geom(const float* __restrict__ pos,
                          const int* __restrict__ src, const int* __restrict__ dst,
                          __bf16* __restrict__ rbf, float* __restrict__ evec)
{
  int idx = blockIdx.x * 256 + threadIdx.x;     // (e, r) pairs
  int e = idx >> 7, r = idx & 127;
  if (e >= EE) return;
  int s = src[e], d = dst[e];
  float rx = pos[s * 3 + 0] - pos[d * 3 + 0];
  float ry = pos[s * 3 + 1] - pos[d * 3 + 1];
  float rz = pos[s * 3 + 2] - pos[d * 3 + 2];
  float dist = sqrtf(rx * rx + ry * ry + rz * rz);
  const float step  = 12.0f / 127.0f;
  const float coeff = -0.5f / (step * step);
  float dd = dist - (float)r * step;
  rbf[(size_t)e * RR + r] = (__bf16)__expf(coeff * dd * dd);
  if (r < 3) {
    float c = (r == 0) ? rx : ((r == 1) ? ry : rz);
    evec[(size_t)e * 3 + r] = c / dist;
  }
}

__global__ void gather_emb(const float* __restrict__ emb, const int* __restrict__ z,
                           float* __restrict__ x)
{
  int i = blockIdx.x * 256 + threadIdx.x;       // N*H
  x[i] = emb[(size_t)z[i >> 8] * HH + (i & 255)];
}

__global__ void zero_f32(float* __restrict__ p, int n)
{
  int i = blockIdx.x * 256 + threadIdx.x;
  if (i < n) p[i] = 0.0f;
}

__global__ void f32_to_bf16(const float* __restrict__ in, __bf16* __restrict__ out, int n)
{
  int i = blockIdx.x * 256 + threadIdx.x;
  if (i < n) out[i] = (__bf16)in[i];
}

// transpose-convert weight: src f32 [K][Nc] -> dst bf16 [Nc][K]
__global__ void wt_transpose(const float* __restrict__ src, __bf16* __restrict__ dst,
                             int K, int Nc)
{
  int i = blockIdx.x * 256 + threadIdx.x;
  if (i >= K * Nc) return;
  int k = i / Nc, c = i % Nc;
  dst[(size_t)c * K + k] = (__bf16)src[i];
}

// layernorm -> bf16 (one wave per row of 256)
__global__ __launch_bounds__(256) void ln_kernel(
    const float* __restrict__ x, const float* __restrict__ w,
    const float* __restrict__ b, __bf16* __restrict__ out)
{
  int wave = threadIdx.x >> 5, lane = threadIdx.x & 31;
  int n = blockIdx.x * 8 + wave;
  const float* xr = x + (size_t)n * HH;
  float vals[8]; float s = 0.f, s2 = 0.f;
#pragma unroll
  for (int k = 0; k < 8; ++k) { float v = xr[lane + k * 32]; vals[k] = v; s += v; s2 += v * v; }
#pragma unroll
  for (int off = 16; off > 0; off >>= 1) {
    s  += __shfl_xor(s,  off, 32);
    s2 += __shfl_xor(s2, off, 32);
  }
  float mu = s * (1.0f / HH);
  float var = s2 * (1.0f / HH) - mu * mu;
  float rstd = rsqrtf(var + 1e-5f);
#pragma unroll
  for (int k = 0; k < 8; ++k) {
    int c = lane + k * 32;
    out[(size_t)n * HH + c] = (__bf16)((vals[k] - mu) * rstd * w[c] + b[c]);
  }
}

// x = (x+dx)/sqrt2 (first N*H);  vec += dvec (all 3*N*H)
__global__ void node_combine(float* __restrict__ x, const float* __restrict__ dx,
                             float* __restrict__ vec, const float* __restrict__ dvec, int nh)
{
  int i = blockIdx.x * 256 + threadIdx.x;
  if (i < nh) x[i] = (x[i] + dx[i]) * INV_SQRT_2f;
  vec[i] += dvec[i];
}

// vec_dot + hcat = [x_bf16 | vnorm_bf16]
__global__ void update_reduce(const float* __restrict__ vp, const float* __restrict__ x,
                              float* __restrict__ vdot, __bf16* __restrict__ hcat)
{
  int i = blockIdx.x * 256 + threadIdx.x;       // N*H
  int n = i >> 8, h = i & 255;
  float dot = 0.f, nrm = 0.f;
#pragma unroll
  for (int d = 0; d < 3; ++d) {
    size_t vb = ((size_t)n * 3 + d) * (2 * HH);
    float v1 = vp[vb + h];
    float v2 = vp[vb + HH + h];
    dot += v1 * v2;
    nrm += v2 * v2;
  }
  vdot[i] = dot * INV_SQRT_Hf;
  hcat[(size_t)n * (2 * HH) + h]      = (__bf16)x[i];
  hcat[(size_t)n * (2 * HH) + HH + h] = (__bf16)sqrtf(nrm + 1e-8f);
}

__global__ void update_final(float* __restrict__ x, float* __restrict__ vec,
                             const float* __restrict__ hsplit, const float* __restrict__ vdot,
                             const float* __restrict__ vp)
{
  int i = blockIdx.x * 256 + threadIdx.x;       // N*H
  int n = i >> 8, h = i & 255;
  const float* hs = hsplit + (size_t)n * (3 * HH);
  float xv1 = hs[h], xv2 = hs[HH + h], xv3 = hs[2 * HH + h];
  float dxv = (xv1 + xv2 * vdot[i]) * INV_SQRT_2f;
  x[i] = (x[i] + dxv) * INV_SQRT_2f;
#pragma unroll
  for (int d = 0; d < 3; ++d) {
    size_t vb = (size_t)n * 3 + d;
    vec[vb * HH + h] += xv3 * vp[vb * (2 * HH) + h];   // vec1 = vp[...,:H]
  }
}

// energy second stage: dot(e1[n,:], oe_w2) + b2, segment-sum by batch
__global__ __launch_bounds__(128) void energy_reduce(
    const __bf16* __restrict__ e1, const float* __restrict__ w2,
    const float* __restrict__ b2, const int* __restrict__ batch,
    float* __restrict__ out)
{
  __shared__ float sm[128];
  int n = blockIdx.x, t = threadIdx.x;
  sm[t] = (float)e1[(size_t)n * 128 + t] * w2[t];
  __syncthreads();
  for (int s = 64; s > 0; s >>= 1) { if (t < s) sm[t] += sm[t + s]; __syncthreads(); }
  if (t == 0) atomicAdd(out + batch[n], sm[0] + b2[0]);
}

__global__ void zero_out64(float* __restrict__ out) { out[threadIdx.x] = 0.0f; }

// =====================================================================
extern "C" void kernel_launch(void* const* d_in, const int* in_sizes, int n_in,
                              void* d_out, int out_size, void* d_ws, size_t ws_size,
                              hipStream_t stream)
{
  const float* pos      = (const float*)d_in[0];
  const float* atom_emb = (const float*)d_in[1];
  const float* ln_w     = (const float*)d_in[2];
  const float* ln_b     = (const float*)d_in[3];
  const float* xp_w1    = (const float*)d_in[4];
  const float* xp_b1    = (const float*)d_in[5];
  const float* xp_w2    = (const float*)d_in[6];
  const float* xp_b2    = (const float*)d_in[7];
  const float* rbf_w    = (const float*)d_in[8];
  const float* rbf_b    = (const float*)d_in[9];
  const float* vec_w    = (const float*)d_in[10];
  const float* xv_w1    = (const float*)d_in[11];
  const float* xv_b1    = (const float*)d_in[12];
  const float* xv_w2    = (const float*)d_in[13];
  const float* xv_b2    = (const float*)d_in[14];
  const float* oe_w1    = (const float*)d_in[15];
  const float* oe_b1    = (const float*)d_in[16];
  const float* oe_w2    = (const float*)d_in[17];
  const float* oe_b2    = (const float*)d_in[18];
  const int*   z        = (const int*)d_in[19];
  const int*   src      = (const int*)d_in[20];
  const int*   dst      = ((const int*)d_in[20]) + EE;   // edge_index [2,E]
  const int*   batch    = (const int*)d_in[21];
  float*       energy   = (float*)d_out;

  // --- bump allocator over workspace ---
  char* wp = (char*)d_ws;
  auto alloc = [&](size_t bytes) -> void* {
    void* r = (void*)wp;
    wp += (bytes + 255) & ~(size_t)255;
    return r;
  };
  __bf16* rbf_bf   = (__bf16*)alloc((size_t)EE * RR * 2);
  float*  evec     = (float*) alloc((size_t)EE * 3 * 4);
  float*  x        = (float*) alloc((size_t)NN * HH * 4);
  float*  vec      = (float*) alloc((size_t)NN * 3 * HH * 4);
  __bf16* xnorm    = (__bf16*)alloc((size_t)NN * HH * 2);
  __bf16* t1       = (__bf16*)alloc((size_t)NN * HH * 2);
  float*  xh       = (float*) alloc((size_t)NN * 3 * HH * 4);
  float*  dx       = (float*) alloc((size_t)NN * HH * 4);
  float*  dvec     = (float*) alloc((size_t)NN * 3 * HH * 4);
  __bf16* vec_bf   = (__bf16*)alloc((size_t)NN * 3 * HH * 2);
  float*  vp       = (float*) alloc((size_t)NN * 3 * 2 * HH * 4);
  float*  vdot     = (float*) alloc((size_t)NN * HH * 4);
  __bf16* hcat     = (__bf16*)alloc((size_t)NN * 2 * HH * 2);
  float*  hsplit   = (float*) alloc((size_t)NN * 3 * HH * 4);
  __bf16* x_bf     = (__bf16*)alloc((size_t)NN * HH * 2);
  __bf16* xp_w1T   = (__bf16*)alloc((size_t)LL * HH * HH * 2);
  __bf16* xp_w2T   = (__bf16*)alloc((size_t)LL * 3 * HH * HH * 2);
  __bf16* rbf_wT   = (__bf16*)alloc((size_t)LL * 3 * HH * RR * 2);
  __bf16* vec_wT   = (__bf16*)alloc((size_t)LL * 2 * HH * HH * 2);
  __bf16* xv_w1T   = (__bf16*)alloc((size_t)LL * HH * 2 * HH * 2);
  __bf16* xv_w2T   = (__bf16*)alloc((size_t)LL * 3 * HH * HH * 2);
  __bf16* oe_w1T   = (__bf16*)alloc((size_t)128 * HH * 2);
  (void)ws_size; (void)n_in; (void)in_sizes; (void)out_size;

  const int NH = NN * HH;

  // --- weight transpose/convert (deterministic, every call) ---
  auto wt = [&](const float* s, __bf16* d, int K, int Nc) {
    wt_transpose<<<(K * Nc + 255) / 256, 256, 0, stream>>>(s, d, K, Nc);
  };
  for (int l = 0; l < LL; ++l) {
    wt(xp_w1 + (size_t)l * HH * HH,       xp_w1T + (size_t)l * HH * HH,       HH,     HH);
    wt(xp_w2 + (size_t)l * HH * 3 * HH,   xp_w2T + (size_t)l * 3 * HH * HH,   HH,     3 * HH);
    wt(rbf_w + (size_t)l * RR * 3 * HH,   rbf_wT + (size_t)l * 3 * HH * RR,   RR,     3 * HH);
    wt(vec_w + (size_t)l * HH * 2 * HH,   vec_wT + (size_t)l * 2 * HH * HH,   HH,     2 * HH);
    wt(xv_w1 + (size_t)l * 2 * HH * HH,   xv_w1T + (size_t)l * HH * 2 * HH,   2 * HH, HH);
    wt(xv_w2 + (size_t)l * HH * 3 * HH,   xv_w2T + (size_t)l * 3 * HH * HH,   HH,     3 * HH);
  }
  wt(oe_w1, oe_w1T, HH, 128);

  // --- edge geometry + RBF (bf16) ---
  edge_geom<<<(EE * RR) / 256, 256, 0, stream>>>(pos, src, dst, rbf_bf, evec);

  // --- init state ---
  gather_emb<<<NH / 256, 256, 0, stream>>>(atom_emb, z, x);
  zero_f32<<<(3 * NH) / 256, 256, 0, stream>>>(vec, 3 * NH);

  auto gemm = [&](const __bf16* A, const __bf16* Bt, const float* bias, void* C,
                  int M, int Nc, int K, int act, int obf) {
    dim3 g(M / 32, (Nc + 255) / 256);
    gemm_wmma_bf16<<<g, 256, 0, stream>>>(A, Bt, bias, C, M, Nc, K, act, obf);
  };

  for (int l = 0; l < LL; ++l) {
    zero_f32<<<NH / 256, 256, 0, stream>>>(dx, NH);
    zero_f32<<<(3 * NH) / 256, 256, 0, stream>>>(dvec, 3 * NH);

    // message phase: xh = silu(LN(x)@W1+b1)@W2+b2
    ln_kernel<<<NN / 8, 256, 0, stream>>>(x, ln_w + (size_t)l * HH, ln_b + (size_t)l * HH, xnorm);
    gemm(xnorm, xp_w1T + (size_t)l * HH * HH, xp_b1 + (size_t)l * HH, t1, NN, HH, HH, 1, 1);
    gemm(t1, xp_w2T + (size_t)l * 3 * HH * HH, xp_b2 + (size_t)l * 3 * HH, xh, NN, 3 * HH, HH, 0, 0);

    // fused rbfh GEMM + message + scatter
    edge_message<<<EE / 16, 256, 0, stream>>>(
        rbf_bf, rbf_wT + (size_t)l * 3 * HH * RR, rbf_b + (size_t)l * 3 * HH,
        xh, vec, evec, src, dst, dx, dvec);

    node_combine<<<(3 * NH) / 256, 256, 0, stream>>>(x, dx, vec, dvec, NH);

    // update phase
    f32_to_bf16<<<(3 * NH) / 256, 256, 0, stream>>>(vec, vec_bf, 3 * NH);
    gemm(vec_bf, vec_wT + (size_t)l * 2 * HH * HH, nullptr, vp, 3 * NN, 2 * HH, HH, 0, 0);
    update_reduce<<<NH / 256, 256, 0, stream>>>(vp, x, vdot, hcat);
    gemm(hcat, xv_w1T + (size_t)l * HH * 2 * HH, xv_b1 + (size_t)l * HH, t1, NN, HH, 2 * HH, 1, 1);
    gemm(t1, xv_w2T + (size_t)l * 3 * HH * HH, xv_b2 + (size_t)l * 3 * HH, hsplit, NN, 3 * HH, HH, 0, 0);
    update_final<<<NH / 256, 256, 0, stream>>>(x, vec, hsplit, vdot, vp);
  }

  // --- energy head ---
  f32_to_bf16<<<NH / 256, 256, 0, stream>>>(x, x_bf, NH);
  gemm(x_bf, oe_w1T, oe_b1, t1, NN, 128, HH, 1, 1);   // e1 bf16 in t1
  zero_out64<<<1, NG, 0, stream>>>(energy);
  energy_reduce<<<NN, 128, 0, stream>>>(t1, oe_w2, oe_b2, batch, energy);
}
